// VectorQuantizer_41721312313826
// MI455X (gfx1250) — matrix operand order repaired and checked by reading, not compiled, not run on hardware
//
#include <hip/hip_runtime.h>
#include <hip/hip_bf16.h>
#include <stdint.h>

// ---------------------------------------------------------------------------
// VQ-VAE vector quantizer for MI455X (gfx1250, wave32, WMMA).
// Distance GEMM via v_wmma_f32_16x16x32_bf16 with bf16 hi/lo split (3 terms)
// for ~fp32 accuracy at bf16-WMMA rates.  Codebook streamed into LDS with
// double-buffered GLOBAL_LOAD_ASYNC_TO_LDS_B128 (ASYNCcnt pipeline).
// ---------------------------------------------------------------------------

#define N_E    1024
#define E_DIM  256
#define NVEC   65536            // 16*64*64
#define ZELEMS 16777216         // 16*256*64*64
#define CCODES 16               // codebook rows per chunk
#define CHUSH  8192             // ushorts per chunk (hi 16x256 | lo 16x256)
#define NCH    (N_E / CCODES)   // 64 chunks

typedef __attribute__((ext_vector_type(16))) __bf16 v16bf;
typedef __attribute__((ext_vector_type(8)))  float  v8f;

union BF16x16 {
    v16bf v;
    unsigned short s[16];
};

// round-to-nearest-even fp32 -> bf16 (bit pattern)
static __device__ __forceinline__ unsigned short f2bf(float f) {
    unsigned int u = __float_as_uint(f);
    u += 0x7FFFu + ((u >> 16) & 1u);
    return (unsigned short)(u >> 16);
}
static __device__ __forceinline__ float bf2f(unsigned short h) {
    return __uint_as_float(((unsigned int)h) << 16);
}

// async 16B global->LDS copy; inst offset applies to BOTH lds and global side
#define ASYNC_CP16(ldsaddr, voff, sbase, imm)                                  \
    asm volatile("global_load_async_to_lds_b128 %0, %1, %2 offset:" #imm       \
                 :: "v"(ldsaddr), "v"(voff), "s"(sbase) : "memory")

// ---------------------------------------------------------------------------
// Kernel 1: codebook prep — bf16 hi/lo split into chunk-major layout
// ( [chunk][ hi 16x256 | lo 16x256 ] ) + exact fp32 row norms.
// Also zero-initializes histogram and loss accumulator.
// ---------------------------------------------------------------------------
__global__ __launch_bounds__(256) void vq_prep_kernel(
    const float* __restrict__ emb,
    unsigned short* __restrict__ eint,
    float* __restrict__ norms,
    unsigned int* __restrict__ counts,
    float* __restrict__ lossacc)
{
    const int j = blockIdx.x;      // code row
    const int t = threadIdx.x;     // dim
    float x = emb[j * E_DIM + t];
    unsigned short h = f2bf(x);
    const int chunkBase = (j >> 4) * CHUSH;
    const int within    = (j & 15) * E_DIM + t;
    eint[chunkBase + within]        = h;                    // hi
    eint[chunkBase + 4096 + within] = f2bf(x - bf2f(h));    // lo

    float s = x * x;
    #pragma unroll
    for (int o = 1; o < 32; o <<= 1) s += __shfl_xor(s, o, 32);
    __shared__ float ps[8];
    if ((t & 31) == 0) ps[t >> 5] = s;
    __syncthreads();
    if (t == 0) {
        float tot = 0.f;
        #pragma unroll
        for (int w = 0; w < 8; ++w) tot += ps[w];
        norms[j] = tot;
        counts[j] = 0u;
        if (j == 0) *lossacc = 0.f;
    }
}

// ---------------------------------------------------------------------------
// Kernel 2: distance GEMM + argmin. One workgroup = 8 waves = 128 z-rows.
// A (z tile) register-resident across full codebook sweep; B (codebook)
// double-buffered through LDS with async global->LDS loads.
// ---------------------------------------------------------------------------
__global__ __launch_bounds__(256) void vq_argmin_kernel(
    const float* __restrict__ z,
    const unsigned short* __restrict__ eint,
    const float* __restrict__ norms,
    int* __restrict__ idx_ws,
    int* __restrict__ idx_out)
{
    __shared__ unsigned short sBuf[2][CHUSH];   // 2 x 16 KB

    const int tid  = threadIdx.x;
    const int wave = tid >> 5;
    const int lane = tid & 31;
    const int l15  = lane & 15;
    const int hs   = lane >> 4;          // which 16-lane half

    const int n0  = blockIdx.x * 128;    // 128 consecutive flat vectors (same b)
    const int row = wave * 16 + l15;     // A-tile row this lane holds (M = l15)
    const int n   = n0 + row;
    const int bb  = n >> 12;             // batch (4096 = 64*64 divides 128 tiles)
    const int hw  = n & 4095;
    const float* zb = z + ((size_t)bb * E_DIM) * 4096 + hw; // z[bb][c][hw]

    // --- software-pipeline prologue: async copy chunk 0 -> buf 0
    const unsigned voff = (unsigned)tid * 16u;
    {
        unsigned lds0 = (unsigned)(size_t)(&sBuf[0][0]) + voff;
        const unsigned short* src = eint;
        ASYNC_CP16(lds0, voff, src, 0);
        ASYNC_CP16(lds0, voff, src, 4096);
        ASYNC_CP16(lds0, voff, src, 8192);
        ASYNC_CP16(lds0, voff, src, 12288);
    }

    // --- Load A tile into registers, bf16 hi/lo split (overlaps with async).
    // A layout (matches B): lanes 0-15 hold K {0..15}, lanes 16-31 hold
    // K {16..31}, element s -> K = 16*half + s.
    BF16x16 a_hi[8], a_lo[8];
    #pragma unroll
    for (int k = 0; k < 8; ++k) {
        #pragma unroll
        for (int j = 0; j < 8; ++j) {
            const int c1 = 32 * k + hs * 16 + j;
            const int c2 = c1 + 8;
            float x1 = zb[(size_t)c1 * 4096];
            float x2 = zb[(size_t)c2 * 4096];
            unsigned short h1 = f2bf(x1);
            a_hi[k].s[j] = h1;
            a_lo[k].s[j] = f2bf(x1 - bf2f(h1));
            unsigned short h2 = f2bf(x2);
            a_hi[k].s[8 + j] = h2;
            a_lo[k].s[8 + j] = f2bf(x2 - bf2f(h2));
        }
    }

    float minv[8];
    int   mini[8];
    #pragma unroll
    for (int r = 0; r < 8; ++r) { minv[r] = 3.4e38f; mini[r] = 0; }

    for (int ch = 0; ch < NCH; ++ch) {
        const int p = ch & 1;
        __syncthreads();     // all waves done computing on buf !p (prev iter)
        if (ch + 1 < NCH) {
            // issue next chunk into the other buffer (overlaps with compute)
            unsigned lds1 = (unsigned)(size_t)(&sBuf[p ^ 1][0]) + voff;
            const unsigned short* src = eint + (size_t)(ch + 1) * CHUSH;
            ASYNC_CP16(lds1, voff, src, 0);
            ASYNC_CP16(lds1, voff, src, 4096);
            ASYNC_CP16(lds1, voff, src, 8192);
            ASYNC_CP16(lds1, voff, src, 12288);
            // 4 outstanding = chunk ch+1; chunk ch copies complete
            asm volatile("s_wait_asynccnt 0x4" ::: "memory");
        } else {
            asm volatile("s_wait_asynccnt 0x0" ::: "memory");
        }
        __syncthreads();     // chunk ch resident in sBuf[p] for all waves

        // B layout: lane half 0 holds K {0..15}, half 1 holds K {16..31},
        // column N = l15 (codebook row, since B^T = E tile row-major).
        const int codeBase = ch * CCODES;
        const unsigned short* pH = &sBuf[p][0] + (size_t)l15 * E_DIM + hs * 16;
        const unsigned short* pL = pH + 4096;

        v8f acc = {};
        #pragma unroll
        for (int k = 0; k < 8; ++k) {
            v16bf bh = *(const v16bf*)(pH + 32 * k);
            v16bf bl = *(const v16bf*)(pL + 32 * k);
            // z.e ~= lo.hi + hi.lo + hi.hi  (all into one f32 accumulator)
            acc = __builtin_amdgcn_wmma_f32_16x16x32_bf16(
                    false, a_lo[k].v, false, bh, (short)0, acc, false, false);
            acc = __builtin_amdgcn_wmma_f32_16x16x32_bf16(
                    false, a_hi[k].v, false, bl, (short)0, acc, false, false);
            acc = __builtin_amdgcn_wmma_f32_16x16x32_bf16(
                    false, a_hi[k].v, false, bh, (short)0, acc, false, false);
        }

        const int   code = codeBase + l15;   // N index for this lane
        const float nrm  = norms[code];
        #pragma unroll
        for (int r = 0; r < 8; ++r) {
            float d = nrm - 2.0f * acc[r];   // ||z||^2 constant -> dropped
            if (d < minv[r]) { minv[r] = d; mini[r] = code; }
        }
    }

    // Cross-lane argmin within each 16-lane half (C layout: VGPR r holds
    // M=r for lanes 0-15 and M=r+8 for lanes 16-31; N = lane&15).
    #pragma unroll
    for (int r = 0; r < 8; ++r) {
        float v = minv[r];
        int   i = mini[r];
        #pragma unroll
        for (int off = 1; off < 16; off <<= 1) {
            float ov = __shfl_xor(v, off, 32);
            int   oi = __shfl_xor(i, off, 32);
            if (ov < v || (ov == v && oi < i)) { v = ov; i = oi; }
        }
        minv[r] = v; mini[r] = i;
    }
    if (l15 == 0) {
        #pragma unroll
        for (int r = 0; r < 8; ++r) {
            const int outn = n0 + wave * 16 + hs * 8 + r;
            idx_ws[outn]  = mini[r];
            idx_out[outn] = mini[r];
        }
    }
}

// ---------------------------------------------------------------------------
// Kernel 3: gather z_q (== straight-through output) + loss accumulation.
// blockIdx.x -> tile of 256 flat vectors, blockIdx.y -> channel c.
// ---------------------------------------------------------------------------
__global__ __launch_bounds__(256) void vq_gather_kernel(
    const float* __restrict__ z,
    const float* __restrict__ emb,
    const int* __restrict__ idx,
    float* __restrict__ zq_out,
    float* __restrict__ lossacc)
{
    const int t  = threadIdx.x;
    const int nn = blockIdx.x * 256 + t;
    const int c  = blockIdx.y;
    const int id = idx[nn];
    const int bb = nn >> 12;
    const int hw = nn & 4095;
    const size_t off = (((size_t)bb * E_DIM + c) << 12) + hw;
    const float zv = z[off];
    const float ev = emb[(size_t)id * E_DIM + c];
    zq_out[off] = ev;            // z_q_st == z_q numerically

    float d = ev - zv;
    float s = d * d;
    #pragma unroll
    for (int o = 1; o < 32; o <<= 1) s += __shfl_xor(s, o, 32);
    __shared__ float ps[8];
    if ((t & 31) == 0) ps[t >> 5] = s;
    __syncthreads();
    if (t == 0) {
        float tot = 0.f;
        #pragma unroll
        for (int w = 0; w < 8; ++w) tot += ps[w];
        atomicAdd(lossacc, tot);
    }
}

// ---------------------------------------------------------------------------
// Kernel 4: code-usage histogram.
// ---------------------------------------------------------------------------
__global__ __launch_bounds__(256) void vq_hist_kernel(
    const int* __restrict__ idx, unsigned int* __restrict__ counts)
{
    const int nn = blockIdx.x * 256 + threadIdx.x;
    atomicAdd(&counts[idx[nn]], 1u);
}

// ---------------------------------------------------------------------------
// Kernel 5: finalize — loss scalar and perplexity.
// ---------------------------------------------------------------------------
__global__ __launch_bounds__(1024) void vq_finalize_kernel(
    const unsigned int* __restrict__ counts,
    const float* __restrict__ lossacc,
    float* __restrict__ out_loss,
    float* __restrict__ out_perp)
{
    const int t = threadIdx.x;                  // 0..1023 == code id
    float em = (float)counts[t] * (1.0f / (float)NVEC);
    float s  = -em * logf(em + 1e-10f);
    #pragma unroll
    for (int o = 1; o < 32; o <<= 1) s += __shfl_xor(s, o, 32);
    __shared__ float ps[32];
    if ((t & 31) == 0) ps[t >> 5] = s;
    __syncthreads();
    if (t == 0) {
        float tot = 0.f;
        #pragma unroll
        for (int w = 0; w < 32; ++w) tot += ps[w];
        *out_perp = expf(tot);
        // both MSE terms equal numerically: loss = (1 + beta) * mean
        *out_loss = (*lossacc) * (1.25f / (float)ZELEMS);
    }
}

// ---------------------------------------------------------------------------
extern "C" void kernel_launch(void* const* d_in, const int* in_sizes, int n_in,
                              void* d_out, int out_size, void* d_ws, size_t ws_size,
                              hipStream_t stream) {
    (void)in_sizes; (void)n_in; (void)out_size; (void)ws_size;
    const float* z   = (const float*)d_in[0];   // [16,256,64,64]
    const float* emb = (const float*)d_in[1];   // [1024,256]

    // d_out tuple layout: z_q_out (16777216 f32) | loss (1) | perplexity (1) | idx (65536 i32)
    float* out_zq   = (float*)d_out;
    float* out_loss = out_zq + ZELEMS;
    float* out_perp = out_zq + ZELEMS + 1;
    int*   out_idx  = (int*)(out_zq + ZELEMS + 2);

    // workspace carve-up
    char* w = (char*)d_ws;
    unsigned short* eint  = (unsigned short*)(w);                 // 1 MB (chunk-major hi|lo)
    float*          norms = (float*)(w + (1024 << 10));           // 4 KB
    unsigned int*   cnts  = (unsigned int*)(w + (1028 << 10));    // 4 KB
    float*          lacc  = (float*)(w + (1032 << 10));           // 16 B
    int*            idxws = (int*)(w + (1032 << 10) + 16);        // 256 KB

    vq_prep_kernel<<<N_E, 256, 0, stream>>>(emb, eint, norms, cnts, lacc);
    vq_argmin_kernel<<<NVEC / 128, 256, 0, stream>>>(z, eint, norms, idxws, out_idx);
    vq_gather_kernel<<<dim3(NVEC / 256, E_DIM), 256, 0, stream>>>(z, emb, idxws, out_zq, lacc);
    vq_hist_kernel<<<NVEC / 256, 256, 0, stream>>>(idxws, cnts);
    vq_finalize_kernel<<<1, 1024, 0, stream>>>(cnts, lacc, out_loss, out_perp);
}
